// MyModule_52699248722435
// MI455X (gfx1250) — matrix operand order repaired and checked by reading, not compiled, not run on hardware
//
#include <hip/hip_runtime.h>
#include <hip/hip_bf16.h>

// ---------------------------------------------------------------------------
// Multi-head attention forward, MI455X (gfx1250), wave32 + WMMA f16->f32.
// B=4, S=2048, E=1024, H=16, D=64.
// ---------------------------------------------------------------------------

typedef __attribute__((ext_vector_type(16))) _Float16 v16h;
typedef __attribute__((ext_vector_type(8)))  _Float16 v8h;
typedef __attribute__((ext_vector_type(8)))  float    v8f;

#define WMMA16(a, b, c) \
  __builtin_amdgcn_wmma_f32_16x16x32_f16(false, (a), false, (b), (short)0, (c), false, false)

static constexpr int Bc = 4;
static constexpr int Sc = 2048;
static constexpr int Ec = 1024;
static constexpr int Hc = 16;
static constexpr int Dc = 64;
static constexpr int HD = Hc * Dc;    // 1024
static constexpr int Mrows = Bc * Sc; // 8192

// ---------------------------------------------------------------------------
// f32 -> f16 flat conversion (for x)
// ---------------------------------------------------------------------------
__global__ void k_cvt_f16(const float* __restrict__ in, _Float16* __restrict__ out, int n) {
  int i = blockIdx.x * blockDim.x + threadIdx.x;
  if (i < n) out[i] = (_Float16)in[i];
}

// ---------------------------------------------------------------------------
// f32 [rows][cols] -> f16 [cols][rows]  (weight transpose so WMMA B-fragments
// are contiguous 32B loads)
// ---------------------------------------------------------------------------
__global__ void k_cvt_transpose_f16(const float* __restrict__ in, _Float16* __restrict__ out,
                                    int rows, int cols) {
  int i = blockIdx.x * blockDim.x + threadIdx.x;
  if (i < rows * cols) {
    int c = i / rows;  // output row (original col)
    int r = i % rows;  // output col (original row)
    out[(size_t)c * rows + r] = (_Float16)in[(size_t)r * cols + c];
  }
}

// ---------------------------------------------------------------------------
// WMMA GEMM: Y = A[MxK] * Bt[NxK]^T + bias, per-wave 32(M)x64(N) tile
// (2 A-fragments x 4 B-fragments = 8 WMMA per 32-wide k-step; each B fragment
//  reused twice -> 24B of load traffic per WMMA).
//   mode 0: store f16, permute to [B,H,S,D]           (Q, K projections)
//   mode 1: store f16, permute to [B,H,D,S] (d-major) (V projection)
//   mode 2: store f32 row-major with ReLU              (output projection)
// ---------------------------------------------------------------------------
__global__ void k_gemm_wmma(const _Float16* __restrict__ A,
                            const _Float16* __restrict__ Bt,
                            const float* __restrict__ bias,
                            void* __restrict__ Out,
                            int M, int N, int K, int mode) {
  const int lane = threadIdx.x & 31;
  const int lp   = lane & 15;
  const int hf   = lane >> 4;
  const int wave = blockIdx.x * (blockDim.x >> 5) + (threadIdx.x >> 5);
  const int ngroups = N >> 6;
  const int mt = wave / ngroups;
  const int ng = wave % ngroups;
  if (mt * 32 >= M) return;
  const int m0 = mt * 32;
  const int n0 = ng * 64;
  const int base0 = hf ? 16 : 0;

  v8f acc[2][4] = {};
  const _Float16* arow0 = A + (size_t)(m0 + lp) * K;
  const _Float16* arow1 = A + (size_t)(m0 + 16 + lp) * K;

  for (int k0 = 0; k0 < K; k0 += 32) {
    // A fragments: lane = row; halfs [base0..base0+7] and [base0+8..base0+15]
    v8h a0lo = *(const v8h*)(arow0 + k0 + base0);
    v8h a0hi = *(const v8h*)(arow0 + k0 + base0 + 8);
    v8h a1lo = *(const v8h*)(arow1 + k0 + base0);
    v8h a1hi = *(const v8h*)(arow1 + k0 + base0 + 8);
    v16h a0, a1;
#pragma unroll
    for (int i = 0; i < 8; ++i) {
      a0[i] = a0lo[i]; a0[i + 8] = a0hi[i];
      a1[i] = a1lo[i]; a1[i + 8] = a1hi[i];
    }
#pragma unroll
    for (int t = 0; t < 4; ++t) {
      // B fragment: lane n = n0+16t+lp, K = k0 + 16*hf + i (contiguous 32B)
      v16h b = *(const v16h*)(Bt + (size_t)(n0 + t * 16 + lp) * K + k0 + 16 * hf);
      acc[0][t] = WMMA16(a0, b, acc[0][t]);
      acc[1][t] = WMMA16(a1, b, acc[1][t]);
    }
  }

#pragma unroll
  for (int u = 0; u < 2; ++u) {
#pragma unroll
    for (int t = 0; t < 4; ++t) {
      const int n = n0 + t * 16 + lp;
      const float bv = bias[n];
#pragma unroll
      for (int r = 0; r < 8; ++r) {
        const int m = m0 + u * 16 + r + 8 * hf;
        float v = acc[u][t][r] + bv;
        if (mode == 2) {
          ((float*)Out)[(size_t)m * N + n] = fmaxf(v, 0.0f);
        } else {
          const int b = m >> 11;        // /S
          const int s = m & (Sc - 1);
          const int h = n >> 6;         // /D
          const int d = n & (Dc - 1);
          size_t idx;
          if (mode == 0) idx = (((size_t)b * Hc + h) * Sc + s) * Dc + d;  // [B,H,S,D]
          else           idx = (((size_t)b * Hc + h) * Dc + d) * Sc + s;  // [B,H,D,S]
          ((_Float16*)Out)[idx] = (_Float16)v;
        }
      }
    }
  }
}

// ---------------------------------------------------------------------------
// Flash attention: one wave = 16 query rows of one (b,h).
// Q,K: [B,H,S,D] f16;  Vt: [B,H,D,S] f16;  mask: [B,S] int;  ctx: [B*S, H*D] f16
// Row sums of P computed via WMMA against an all-ones B matrix (replaces 32
// cross-lane shuffles per kv-block with one XDL op, co-executing with VALU).
// ---------------------------------------------------------------------------
__global__ void k_attn_wmma(const _Float16* __restrict__ Q,
                            const _Float16* __restrict__ Km,
                            const _Float16* __restrict__ Vt,
                            const int* __restrict__ mask,
                            _Float16* __restrict__ ctx) {
  __shared__ _Float16 pbuf[8][16][32];  // per-wave P tile (C-layout -> A-layout bridge)

  const int lane = threadIdx.x & 31;
  const int lp   = lane & 15;
  const int hf   = lane >> 4;
  const int w    = threadIdx.x >> 5;
  const int wave = blockIdx.x * 8 + w;
  const int bh   = wave >> 7;          // S/16 = 128 q-tiles per head
  const int qt   = wave & 127;
  const int b    = bh >> 4;
  const int h    = bh & 15;
  const int q0   = qt * 16;
  const int base0 = hf ? 16 : 0;
  const float scale = 0.125f;          // 1/sqrt(64)
  const float NEG = -1.0e9f;

  v16h ones;
#pragma unroll
  for (int i = 0; i < 16; ++i) ones[i] = (_Float16)1.0f;

  // Q A-fragments (d 0..31 and d 32..63), loaded once
  const _Float16* qrow = Q + ((size_t)bh * Sc + q0 + lp) * Dc;
  v16h qa0, qa1;
  {
    v8h l0 = *(const v8h*)(qrow + base0);
    v8h h0 = *(const v8h*)(qrow + base0 + 8);
    v8h l1 = *(const v8h*)(qrow + 32 + base0);
    v8h h1 = *(const v8h*)(qrow + 32 + base0 + 8);
#pragma unroll
    for (int i = 0; i < 8; ++i) {
      qa0[i] = l0[i]; qa0[i + 8] = h0[i];
      qa1[i] = l1[i]; qa1[i + 8] = h1[i];
    }
  }

  const _Float16* Kbase = Km + (size_t)bh * Sc * Dc;
  const _Float16* Vbase = Vt + (size_t)bh * Dc * Sc;
  const int* mbase = mask + b * Sc;

  v8f acc[4] = {};
  float mrow[8], srow[8];
#pragma unroll
  for (int r = 0; r < 8; ++r) { mrow[r] = -3.0e38f; srow[r] = 0.0f; }

  for (int kb = 0; kb < Sc; kb += 32) {
    // ---- scores: S = Q @ K^T (two 16-col tiles) ----
    const _Float16* krow0 = Kbase + (size_t)(kb + lp) * Dc + 16 * hf;
    const _Float16* krow1 = Kbase + (size_t)(kb + 16 + lp) * Dc + 16 * hf;
    v16h kb0a = *(const v16h*)(krow0);
    v16h kb0b = *(const v16h*)(krow0 + 32);
    v16h kb1a = *(const v16h*)(krow1);
    v16h kb1b = *(const v16h*)(krow1 + 32);
    if (kb + 32 < Sc) {
      __builtin_prefetch(krow0 + 32 * Dc, 0, 0);  // next K block -> global_prefetch_b8
      __builtin_prefetch(Vbase + (size_t)lp * Sc + kb + 32, 0, 0);
    }

    v8f s0 = {}, s1 = {};
    s0 = WMMA16(qa0, kb0a, s0);
    s0 = WMMA16(qa1, kb0b, s0);
    s1 = WMMA16(qa0, kb1a, s1);
    s1 = WMMA16(qa1, kb1b, s1);

    const int mk0 = mbase[kb + lp];
    const int mk1 = mbase[kb + 16 + lp];

    // ---- online softmax (row = r + 8*hf lives across one 16-lane half) ----
#pragma unroll
    for (int r = 0; r < 8; ++r) {
      float x0 = mk0 ? s0[r] * scale : NEG;
      float x1 = mk1 ? s1[r] * scale : NEG;
      float mx = fmaxf(x0, x1);
      mx = fmaxf(mx, __shfl_xor(mx, 1));
      mx = fmaxf(mx, __shfl_xor(mx, 2));
      mx = fmaxf(mx, __shfl_xor(mx, 4));
      mx = fmaxf(mx, __shfl_xor(mx, 8));
      float mnew = fmaxf(mrow[r], mx);
      float alpha = __expf(mrow[r] - mnew);
      float p0 = __expf(x0 - mnew);
      float p1 = __expf(x1 - mnew);
      srow[r] *= alpha;
      mrow[r] = mnew;
      pbuf[w][r + 8 * hf][lp]      = (_Float16)p0;
      pbuf[w][r + 8 * hf][16 + lp] = (_Float16)p1;
#pragma unroll
      for (int t = 0; t < 4; ++t) acc[t][r] *= alpha;
    }

    // ---- P tile: C-layout -> A-layout via LDS (same wave; dscnt-ordered) ----
    v8h plo = *(const v8h*)&pbuf[w][lp][base0];
    v8h phi = *(const v8h*)&pbuf[w][lp][base0 + 8];
    v16h pa;
#pragma unroll
    for (int i = 0; i < 8; ++i) { pa[i] = plo[i]; pa[i + 8] = phi[i]; }

    // ---- row sums of P via WMMA against all-ones B (replicated per column) ----
    v8f ts = {};
    ts = WMMA16(pa, ones, ts);
#pragma unroll
    for (int r = 0; r < 8; ++r) srow[r] += ts[r];

    // ---- O += P @ V : V B-fragment is a contiguous 32B row of Vt ----
#pragma unroll
    for (int t = 0; t < 4; ++t) {
      v16h vb = *(const v16h*)(Vbase + (size_t)(t * 16 + lp) * Sc + kb + 16 * hf);
      acc[t] = WMMA16(pa, vb, acc[t]);
    }
  }

  // ---- normalize and store ctx as [B*S, H*D] f16 row-major ----
  float inv[8];
#pragma unroll
  for (int r = 0; r < 8; ++r) inv[r] = 1.0f / srow[r];
#pragma unroll
  for (int t = 0; t < 4; ++t) {
#pragma unroll
    for (int r = 0; r < 8; ++r) {
      const int srowg = q0 + r + 8 * hf;
      const int col = h * Dc + t * 16 + lp;
      ctx[((size_t)b * Sc + srowg) * HD + col] = (_Float16)(acc[t][r] * inv[r]);
    }
  }
}

// ---------------------------------------------------------------------------
// Host-side launch
// ---------------------------------------------------------------------------
extern "C" void kernel_launch(void* const* d_in, const int* in_sizes, int n_in,
                              void* d_out, int out_size, void* d_ws, size_t ws_size,
                              hipStream_t stream) {
  const float* x  = (const float*)d_in[0];
  const int*   mk = (const int*)d_in[1];
  const float* Wq = (const float*)d_in[2];
  const float* bq = (const float*)d_in[3];
  const float* Wk = (const float*)d_in[4];
  const float* bk = (const float*)d_in[5];
  const float* Wv = (const float*)d_in[6];
  const float* bv = (const float*)d_in[7];
  const float* Wo = (const float*)d_in[8];
  const float* bo = (const float*)d_in[9];
  float* out = (float*)d_out;

  char* ws = (char*)d_ws;
  const size_t MB = 1u << 20;
  _Float16* xh  = (_Float16*)(ws + 0);         // 16 MB  [8192 x 1024]
  _Float16* wqt = (_Float16*)(ws + 16 * MB);   //  2 MB  [1024 x 1024] transposed
  _Float16* wkt = (_Float16*)(ws + 18 * MB);
  _Float16* wvt = (_Float16*)(ws + 20 * MB);
  _Float16* wot = (_Float16*)(ws + 22 * MB);
  _Float16* Qh  = (_Float16*)(ws + 24 * MB);   // 16 MB  [B,H,S,D]
  _Float16* Kh  = (_Float16*)(ws + 40 * MB);   // 16 MB  [B,H,S,D]
  _Float16* Vh  = (_Float16*)(ws + 56 * MB);   // 16 MB  [B,H,D,S]
  _Float16* Ch  = (_Float16*)(ws + 72 * MB);   // 16 MB  [8192 x 1024]

  // 1) precision conversion + weight transposes
  {
    int n = Mrows * Ec;
    k_cvt_f16<<<(n + 255) / 256, 256, 0, stream>>>(x, xh, n);
    int nw = Ec * HD;
    k_cvt_transpose_f16<<<(nw + 255) / 256, 256, 0, stream>>>(Wq, wqt, Ec, HD);
    k_cvt_transpose_f16<<<(nw + 255) / 256, 256, 0, stream>>>(Wk, wkt, Ec, HD);
    k_cvt_transpose_f16<<<(nw + 255) / 256, 256, 0, stream>>>(Wv, wvt, Ec, HD);
    k_cvt_transpose_f16<<<(nw + 255) / 256, 256, 0, stream>>>(Wo, wot, HD, Ec);
  }

  // 2) Q/K/V projections: waves = (M/32)*(N/64) = 4096, 8 waves/block
  {
    dim3 grid(512), block(256);
    k_gemm_wmma<<<grid, block, 0, stream>>>(xh, wqt, bq, Qh, Mrows, HD, Ec, 0);
    k_gemm_wmma<<<grid, block, 0, stream>>>(xh, wkt, bk, Kh, Mrows, HD, Ec, 0);
    k_gemm_wmma<<<grid, block, 0, stream>>>(xh, wvt, bv, Vh, Mrows, HD, Ec, 1);
  }

  // 3) attention: waves = B*H*(S/16) = 8192, 8 waves/block
  k_attn_wmma<<<1024, 256, 0, stream>>>(Qh, Kh, Vh, mk, Ch);

  // 4) output projection + bias + ReLU -> f32
  k_gemm_wmma<<<512, 256, 0, stream>>>(Ch, wot, bo, out, Mrows, Ec, HD, 2);
}
